// RelationalGCN_73323681677520
// MI455X (gfx1250) — matrix operand order repaired and checked by reading, not compiled, not run on hardware
//
#include <hip/hip_runtime.h>
#include <hip/hip_bf16.h>
#include <stdint.h>

// ---------------------------------------------------------------------------
// R-GCN (5 layers) for MI455X / gfx1250.
//  - All matmuls via v_wmma_f32_16x16x32_bf16 (f32 accumulate), 4 WMMA/wave/K-step.
//  - Edge scatter-mean via native global_atomic_add_f32 (inline asm).
//  - Workspace: inv_cnt[N*12] + hA[N*128] + hB[N*128] + tmp[N*128] + offs[64]
//    ~= 79.2 MB.
// ---------------------------------------------------------------------------

typedef __bf16 v16bf __attribute__((ext_vector_type(16)));
typedef float  v8f   __attribute__((ext_vector_type(8)));

#define DH     128
#define NREL   12
#define TILE_M 64
#define AS_STRIDE 40    // 32 + 8 pad: 80B row stride, 16B aligned, conflict-free
#define BS_STRIDE 40    // same for transposed B

static __device__ __forceinline__ __bf16 f2bf(float f) {
  return static_cast<__bf16>(f);     // native v_cvt to bf16 (RNE)
}

static __device__ __forceinline__ void atomic_add_f32g(float* p, float v) {
  // Native CDNA5 FP32 global atomic (no CAS loop). No return -> STOREcnt.
  asm volatile("global_atomic_add_f32 %0, %1, off" :: "v"(p), "v"(v) : "memory");
}

// ---------------------------------------------------------------------------
// C[M,128] = A[M,K] @ B[K,128] (+ bias), bf16 WMMA with f32 accumulation.
// Block: 256 threads = 8 waves; block tile M=64, N=128.
// Wave w: M-strip = (w>>1)*16, N-group = (w&1)*64 (four 16x16 WMMA tiles).
// Per K-step per wave: 1 A fragment, 4 B fragments, 4 WMMAs.
// ---------------------------------------------------------------------------
__global__ __launch_bounds__(256) void gemm_bf16_wmma(
    const float* __restrict__ A, const float* __restrict__ B,
    const float* __restrict__ bias, float* __restrict__ C, int M, int K)
{
  __shared__ alignas(16) __bf16 As[TILE_M * AS_STRIDE];  // [64 rows][32 k]
  __shared__ alignas(16) __bf16 Bs[DH * BS_STRIDE];      // transposed [128 n][32 k]

  const int t    = threadIdx.x;
  const int wave = t >> 5;
  const int lane = t & 31;
  const int l16  = lane & 15;
  const int hi   = lane >> 4;             // 0: lanes 0-15, 1: lanes 16-31
  const int row0 = blockIdx.x * TILE_M;
  const int mt   = wave >> 1;             // 0..3 : M strip
  const int ng   = (wave & 1) * 64;       // 0 or 64 : N group base

  v8f c0 = {}, c1 = {}, c2 = {}, c3 = {};

  for (int k0 = 0; k0 < K; k0 += 32) {
    // ---- stage A tile: 64 rows x 32 k (two float4 per thread) ----
    #pragma unroll
    for (int i = 0; i < 2; ++i) {
      int q  = t + (i << 8);              // 0..511 float4 index
      int r  = q >> 3;                    // row 0..63
      int c4 = (q & 7) << 2;              // k-col 0,4,..,28
      int rg = row0 + r; if (rg >= M) rg = M - 1;   // clamp: safe read, store guarded
      const float* ap = A + (size_t)rg * K + k0 + c4;
      const float4 v = *(const float4*)ap;
      if (k0 + 32 < K) __builtin_prefetch(ap + 32, 0, 1);  // global_prefetch next k-tile
      __bf16* d = As + r * AS_STRIDE + c4;
      d[0] = f2bf(v.x); d[1] = f2bf(v.y); d[2] = f2bf(v.z); d[3] = f2bf(v.w);
    }
    // ---- stage B tile transposed: 32 k x 128 n (4 float4 per thread) ----
    #pragma unroll
    for (int i = 0; i < 4; ++i) {
      int q  = t + (i << 8);              // 0..1023 float4 index
      int kk = q >> 5;                    // k 0..31
      int n  = (q & 31) << 2;             // n 0,4,..,124
      const float4 v = *(const float4*)(B + (size_t)(k0 + kk) * DH + n);
      Bs[(n + 0) * BS_STRIDE + kk] = f2bf(v.x);
      Bs[(n + 1) * BS_STRIDE + kk] = f2bf(v.y);
      Bs[(n + 2) * BS_STRIDE + kk] = f2bf(v.z);
      Bs[(n + 3) * BS_STRIDE + kk] = f2bf(v.w);
    }
    __syncthreads();

    // ---- A fragment (ISA 7.12.2, 16-bit A 16x32): lane row = l16,
    //      K = hi*8 + j (j<8) and hi*8 + 16 + j (j>=8); both spans 16B aligned ----
    v16bf a;
    {
      const __bf16* ap = As + (mt * 16 + l16) * AS_STRIDE + hi * 8;
      #pragma unroll
      for (int j = 0; j < 8; ++j) { a[j] = ap[j]; a[j + 8] = ap[16 + j]; }
    }
    // ---- B fragments: lane col n = tile + l16, K = hi*8 + j (j<8) and hi*8 + 16 + j (j>=8) ----
    v16bf b0, b1, b2, b3;
    {
      const __bf16* bp = Bs + (ng + l16) * BS_STRIDE + hi * 8;
      #pragma unroll
      for (int j = 0; j < 8; ++j) {
        b0[j] = bp[j];                  b0[j + 8] = bp[16 + j];
        b1[j] = bp[16 * BS_STRIDE + j]; b1[j + 8] = bp[16 * BS_STRIDE + 16 + j];
        b2[j] = bp[32 * BS_STRIDE + j]; b2[j + 8] = bp[32 * BS_STRIDE + 16 + j];
        b3[j] = bp[48 * BS_STRIDE + j]; b3[j + 8] = bp[48 * BS_STRIDE + 16 + j];
      }
    }

    c0 = __builtin_amdgcn_wmma_f32_16x16x32_bf16(false, a, false, b0, (short)0, c0, false, false);
    c1 = __builtin_amdgcn_wmma_f32_16x16x32_bf16(false, a, false, b1, (short)0, c1, false, false);
    c2 = __builtin_amdgcn_wmma_f32_16x16x32_bf16(false, a, false, b2, (short)0, c2, false, false);
    c3 = __builtin_amdgcn_wmma_f32_16x16x32_bf16(false, a, false, b3, (short)0, c3, false, false);
    __syncthreads();
  }

  // ---- epilogue: C/D layout lane col = l16, row = v + hi*8 ----
  const int mbase = row0 + mt * 16 + hi * 8;
  const int ncol  = ng + l16;
  #pragma unroll
  for (int v = 0; v < 8; ++v) {
    int row = mbase + v;
    if (row < M) {
      float x0 = c0[v], x1 = c1[v], x2 = c2[v], x3 = c3[v];
      if (bias) {
        x0 += bias[ncol];      x1 += bias[ncol + 16];
        x2 += bias[ncol + 32]; x3 += bias[ncol + 48];
      }
      float* cp = C + (size_t)row * DH + ncol;
      cp[0] = x0; cp[16] = x1; cp[32] = x2; cp[48] = x3;
    }
  }
}

// ---------------------------------------------------------------------------
// Per-(dst,relation) in-degree counts, then inverse (mean denominator).
// ---------------------------------------------------------------------------
__global__ void count_edges(const int* __restrict__ dst, const int* __restrict__ etype,
                            int E, int* __restrict__ cnt) {
  int e = blockIdx.x * blockDim.x + threadIdx.x;
  if (e < E) atomicAdd(&cnt[dst[e] * NREL + etype[e]], 1);
}

__global__ void make_inv(const int* __restrict__ cnt, float* __restrict__ inv, int n) {
  int i = blockIdx.x * blockDim.x + threadIdx.x;
  if (i < n) { int c = cnt[i]; inv[i] = 1.0f / (float)(c > 1 ? c : 1); }
}

// ---------------------------------------------------------------------------
// Scatter-mean for one relation: out[dst] += tmp[src] * inv[dst,rel].
// One wave per edge (32 lanes x 4 floats = 128); 4 edges per wave-loop,
// uniform per-wave predicates.
// ---------------------------------------------------------------------------
__global__ __launch_bounds__(256) void scatter_mean(
    const int* __restrict__ src, const int* __restrict__ dst,
    const int* __restrict__ etype, int E, int rel,
    const float* __restrict__ tmp, const float* __restrict__ inv,
    float* __restrict__ out)
{
  int wave = threadIdx.x >> 5;
  int lane = threadIdx.x & 31;
  int ebase = blockIdx.x * 32 + wave * 4;
  for (int i = 0; i < 4; ++i) {
    int e = ebase + i;
    if (e < E && etype[e] == rel) {
      int s = src[e], d = dst[e];
      float wgt = inv[d * NREL + rel];
      const float4 m = *(const float4*)(tmp + (size_t)s * DH + lane * 4);
      float* o = out + (size_t)d * DH + lane * 4;
      atomic_add_f32g(o + 0, m.x * wgt);
      atomic_add_f32g(o + 1, m.y * wgt);
      atomic_add_f32g(o + 2, m.z * wgt);
      atomic_add_f32g(o + 3, m.w * wgt);
    }
  }
}

__global__ void relu_kernel(float* __restrict__ h, int n) {
  int i = blockIdx.x * blockDim.x + threadIdx.x;
  if (i < n) { float v = h[i]; h[i] = v > 0.f ? v : 0.f; }
}

// offsets[g] = #nodes with batch < g  (batch is sorted) -> binary search.
__global__ void graph_offsets(const int* __restrict__ batch, int Nn, int G,
                              int* __restrict__ offs) {
  int g = threadIdx.x;
  if (g >= G) return;
  int lo = 0, hi = Nn;
  while (lo < hi) { int mid = (lo + hi) >> 1; if (batch[mid] < g) lo = mid + 1; else hi = mid; }
  offs[g] = lo;
}

__global__ void gather_out(const float* __restrict__ h, const int* __restrict__ offs,
                           const int* __restrict__ ihead, const int* __restrict__ itail,
                           float* __restrict__ out) {
  int g = blockIdx.x, t = threadIdx.x;
  int node = (t < DH) ? (ihead[g] + offs[g]) : (itail[g] + offs[g]);
  int c = t & (DH - 1);
  out[(size_t)g * 2 * DH + t] = h[(size_t)node * DH + c];
}

// ---------------------------------------------------------------------------
extern "C" void kernel_launch(void* const* d_in, const int* in_sizes, int n_in,
                              void* d_out, int out_size, void* d_ws, size_t ws_size,
                              hipStream_t stream) {
  const float* x     = (const float*)d_in[0];
  const int*   eidx  = (const int*)d_in[1];
  const int*   etype = (const int*)d_in[2];
  const int*   batch = (const int*)d_in[3];
  const int*   ihead = (const int*)d_in[4];
  const int*   itail = (const int*)d_in[5];

  const int Nn = in_sizes[0] / 768;   // 50000
  const int E  = in_sizes[1] / 2;     // 800000
  const int G  = in_sizes[4];         // 64
  const int* src = eidx;
  const int* dst = eidx + E;

  // Workspace carve-up (~79.2 MB; all regions 16B aligned).
  char* w = (char*)d_ws;
  float* inv = (float*)w;  w += (size_t)Nn * NREL * sizeof(float);
  float* hA  = (float*)w;  w += (size_t)Nn * DH * sizeof(float);
  float* hB  = (float*)w;  w += (size_t)Nn * DH * sizeof(float);
  float* tmp = (float*)w;  w += (size_t)Nn * DH * sizeof(float);
  int*   offs = (int*)w;
  (void)ws_size; (void)n_in;

  // Degree counts (layer-invariant) -> inverse means.
  hipMemsetAsync(inv, 0, (size_t)Nn * NREL * sizeof(float), stream);
  count_edges<<<(E + 255) / 256, 256, 0, stream>>>(dst, etype, E, (int*)inv);
  make_inv<<<(Nn * NREL + 255) / 256, 256, 0, stream>>>((const int*)inv, inv, Nn * NREL);

  const float* cur = nullptr;
  float* nxt = hA;
  const int gblocks = (Nn + TILE_M - 1) / TILE_M;
  const int sblocks = (E + 31) / 32;

  for (int l = 0; l < 5; ++l) {
    const float* Wl = (const float*)d_in[6 + 3 * l];   // [12, K, 128]
    const float* Rl = (const float*)d_in[7 + 3 * l];   // [K, 128]
    const float* Bl = (const float*)d_in[8 + 3 * l];   // [128]
    const float* Ain = (l == 0) ? x : cur;
    const int K = (l == 0) ? 768 : DH;

    // out = Ain @ W_root + b
    gemm_bf16_wmma<<<gblocks, 256, 0, stream>>>(Ain, Rl, Bl, nxt, Nn, K);

    // out += sum_r scatter_mean(Ain @ W[r])
    for (int r = 0; r < NREL; ++r) {
      gemm_bf16_wmma<<<gblocks, 256, 0, stream>>>(
          Ain, Wl + (size_t)r * K * DH, nullptr, tmp, Nn, K);
      scatter_mean<<<sblocks, 256, 0, stream>>>(src, dst, etype, E, r, tmp, inv, nxt);
    }

    if (l < 4) relu_kernel<<<(Nn * DH + 255) / 256, 256, 0, stream>>>(nxt, Nn * DH);

    cur = nxt;
    nxt = (nxt == hA) ? hB : hA;
  }

  // Per-graph offsets (batch sorted) + head/tail gather -> [G, 256].
  graph_offsets<<<1, 64, 0, stream>>>(batch, Nn, G, offs);
  gather_out<<<G, 256, 0, stream>>>(cur, offs, ihead, itail, (float*)d_out);
}
